// RoIHead_2585570312365
// MI455X (gfx1250) — matrix operand order repaired and checked by reading, not compiled, not run on hardware
//
#include <hip/hip_runtime.h>
#include <hip/hip_bf16.h>

typedef __attribute__((ext_vector_type(16))) _Float16 v16h;
typedef __attribute__((ext_vector_type(8)))  float    v8f;
typedef __attribute__((ext_vector_type(4)))  unsigned v4u;
typedef __attribute__((ext_vector_type(4)))  int      v4i;
typedef __attribute__((ext_vector_type(8)))  int      v8i;

#define N_ROIS   2000
#define M_PAD    2048
#define FH       50
#define FW       50
#define CCH      512
#define PH       7
#define PW       7
#define FC_IN    25088      // 7*7*512; /64 = 392 k-steps
#define FC_OUT   4096       // /64 = 64 k-steps
#define N_LOC    84
#define N_SCORE  21
#define NH       105
#define NH_PAD   128

#define SZ_A   ((size_t)M_PAD * FC_IN * 2)
#define SZ_WT  ((size_t)FC_OUT * FC_IN * 2)
#define SZ_FC  ((size_t)M_PAD * FC_OUT * 2)
#define SZ_WH  ((size_t)NH_PAD * FC_OUT * 2)

#if __has_builtin(__builtin_amdgcn_tensor_load_to_lds) && __has_builtin(__builtin_amdgcn_s_wait_tensorcnt)
#define USE_TDM 1
#else
#define USE_TDM 0
#endif

// ============================================================
// 1) RoI align (bilinear crop_and_resize), write f16 A matrix
// ============================================================
__global__ __launch_bounds__(256) void pool_kernel(
    const float* __restrict__ feature, const float* __restrict__ rois,
    const int* __restrict__ ihp, const int* __restrict__ iwp,
    _Float16* __restrict__ A)
{
    const int n   = blockIdx.x;
    const int tid = threadIdx.x;
    _Float16* out = A + (size_t)n * FC_IN;

    if (n >= N_ROIS) {
        for (int i = tid; i < FC_IN; i += 256) out[i] = (_Float16)0.f;
        return;
    }

    const float imgH = (float)(*ihp);
    const float imgW = (float)(*iwp);
    const float y1 = rois[n * 4 + 0] / imgH;
    const float x1 = rois[n * 4 + 1] / imgW;
    const float y2 = rois[n * 4 + 2] / imgH;
    const float x2 = rois[n * 4 + 3] / imgW;

    const float sy = (y2 - y1) * (float)(FH - 1) / (float)(PH - 1);
    const float sx = (x2 - x1) * (float)(FW - 1) / (float)(PW - 1);

    for (int p = 0; p < PH * PW; ++p) {
        const int py = p / PW, px = p % PW;
        const float iny = y1 * (float)(FH - 1) + (float)py * sy;
        const float inx = x1 * (float)(FW - 1) + (float)px * sx;
        const bool valid = (iny >= 0.f) && (iny <= (float)(FH - 1)) &&
                           (inx >= 0.f) && (inx <= (float)(FW - 1));
        const float y0f = floorf(iny), x0f = floorf(inx);
        const float yl = iny - y0f, xl = inx - x0f;
        const int y0  = min(max((int)y0f, 0), FH - 1);
        const int y1i = min(max((int)y0f + 1, 0), FH - 1);
        const int x0  = min(max((int)x0f, 0), FW - 1);
        const int x1i = min(max((int)x0f + 1, 0), FW - 1);

        const float* p00 = feature + ((size_t)y0  * FW + x0 ) * CCH;
        const float* p01 = feature + ((size_t)y0  * FW + x1i) * CCH;
        const float* p10 = feature + ((size_t)y1i * FW + x0 ) * CCH;
        const float* p11 = feature + ((size_t)y1i * FW + x1i) * CCH;

        _Float16* dst = out + (size_t)p * CCH;
        for (int c = tid; c < CCH; c += 256) {
            float top = p00[c] * (1.f - xl) + p01[c] * xl;
            float bot = p10[c] * (1.f - xl) + p11[c] * xl;
            float v   = top * (1.f - yl) + bot * yl;
            dst[c] = valid ? (_Float16)v : (_Float16)0.f;
        }
    }
}

// ============================================================
// 2) W_fc fp32 [K, N] -> f16 W^T [N, K] (LDS tiled transpose)
// ============================================================
__global__ __launch_bounds__(256) void convert_wfc(
    const float* __restrict__ W, _Float16* __restrict__ Wt)
{
    __shared__ float tile[32][33];
    const int tx = threadIdx.x;
    const int ty = threadIdx.y;
    const int nb = blockIdx.x * 32;
    const size_t kb = (size_t)blockIdx.y * 32;

#pragma unroll
    for (int i = 0; i < 4; ++i) {
        int k = ty + i * 8;
        tile[k][tx] = W[(kb + k) * FC_OUT + nb + tx];
    }
    __syncthreads();
#pragma unroll
    for (int i = 0; i < 4; ++i) {
        int nn = ty + i * 8;
        Wt[(size_t)(nb + nn) * FC_IN + kb + tx] = (_Float16)tile[tx][nn];
    }
}

// ============================================================
// 3) Pack head weights^T, zero padded to [128, 4096]
// ============================================================
__global__ __launch_bounds__(256) void prep_wh(
    const float* __restrict__ Wl, const float* __restrict__ Ws,
    _Float16* __restrict__ Wh)
{
    const int r = blockIdx.x;
    for (int k = threadIdx.x; k < FC_OUT; k += 256) {
        float v = 0.f;
        if (r < N_LOC)      v = Wl[(size_t)k * N_LOC + r];
        else if (r < NH)    v = Ws[(size_t)k * N_SCORE + (r - N_LOC)];
        Wh[(size_t)r * FC_OUT + k] = (_Float16)v;
    }
}

// ============================================================
// TDM helpers: 2D f16 tile load Global -> LDS (one instruction)
// ============================================================
#if USE_TDM
typedef __attribute__((address_space(3))) const void* lds_cptr_t;

__device__ __forceinline__ unsigned lds_addr_of(const void* p) {
    return (unsigned)(size_t)(lds_cptr_t)p;
}

__device__ __forceinline__ void tdm_load_tile_f16(
    unsigned lds_byte_addr, const _Float16* gsrc,
    unsigned tile_w, unsigned tile_h,          // elements / rows
    unsigned tensor_w, unsigned stride_elems)  // dim0 extent, dim0 stride
{
    const unsigned long long ga = (unsigned long long)(size_t)gsrc;
    const unsigned tensor_h = 1u << 20;        // tile never clips in dim1

    v4u g0;
    g0.x = 1u;                                             // count=1, user D#
    g0.y = lds_byte_addr;                                  // lds_addr
    g0.z = (unsigned)ga;                                   // global_addr lo
    g0.w = ((unsigned)(ga >> 32) & 0x01FFFFFFu)            // global_addr[56:32]
           | 0x80000000u;                                  // type = 2 ("image")

    v8i g1;
    g1[0] = (int)(1u << 16);                               // data_size = 1 (2B)
    g1[1] = (int)((tensor_w & 0xFFFFu) << 16);             // tensor_dim0 lo16
    g1[2] = (int)(((tensor_w >> 16) & 0xFFFFu) |
                  ((tensor_h & 0xFFFFu) << 16));           // dim0 hi16 | dim1 lo16
    g1[3] = (int)(((tensor_h >> 16) & 0xFFFFu) |
                  ((tile_w & 0xFFFFu) << 16));             // dim1 hi16 | tile_dim0
    g1[4] = (int)(tile_h & 0xFFFFu);                       // tile_dim1 (tile_dim2=0)
    g1[5] = (int)stride_elems;                             // tensor_dim0_stride lo32
    g1[6] = 0;                                             // stride hi / dim1_stride
    g1[7] = 0;

    v4i z = (v4i)0;
#if defined(__clang_major__) && (__clang_major__ >= 23)
    __builtin_amdgcn_tensor_load_to_lds(g0, g1, z, z, (v8i)0, 0);
#else
    __builtin_amdgcn_tensor_load_to_lds(g0, g1, z, z, 0);
#endif
}
#endif // USE_TDM

// ============================================================
// 4) WMMA GEMM: C[M,N] = A[M,K] * B^T[N,K], f16 in / f32 acc
//    BM=128 BN=128 BK=64, 8 waves, wave tile 32x64 (16 WMMA/step)
//    TDM path: async DMA tiles into LDS, 1 barrier per k-step.
// ============================================================
template <int EPI>
__global__ __launch_bounds__(256) void gemm_f16(
    const _Float16* __restrict__ Amat, const _Float16* __restrict__ Bt, int K,
    const float* __restrict__ bias0, const float* __restrict__ bias1,
    _Float16* __restrict__ outH,
    float* __restrict__ outLoc, float* __restrict__ outScore)
{
    constexpr int BM = 128, BN = 128, BK = 64;
    __shared__ _Float16 As[2][BM][BK];   // 2 x 16 KB
    __shared__ _Float16 Bs[2][BN][BK];   // 2 x 16 KB

    const int tid  = threadIdx.x;
    const int lane = tid & 31;
    const int wid  = tid >> 5;
    const int wm   = (wid & 3) * 32;
    const int wn   = (wid >> 2) * 64;

    const size_t m0 = (size_t)blockIdx.x * BM;
    const size_t n0 = (size_t)blockIdx.y * BN;
    const _Float16* Ag = Amat + m0 * (size_t)K;
    const _Float16* Bg = Bt   + n0 * (size_t)K;

    v8f acc[2][4] = {};

    const int steps = K / BK;
    const int fr  = lane & 15;
    const int fhA = (lane >> 4) * 8;    // A frag: K {fhA..+7} and {fhA+16..+23}
    const int fhB = (lane >> 4) * 8;    // B frag: K {fhB..+7} and {fhB+16..+23}

#if !USE_TDM
    const int r0 = tid >> 3;            // 0..31
    const int c0 = (tid & 7) * 8;       // 16B segment within 128B row
    uint4 ra[4], rb[4];
    {
#pragma unroll
        for (int p = 0; p < 4; ++p) {
            const size_t off = (size_t)(r0 + p * 32) * K + c0;
            ra[p] = *(const uint4*)(Ag + off);
            rb[p] = *(const uint4*)(Bg + off);
        }
#pragma unroll
        for (int p = 0; p < 4; ++p) {
            *(uint4*)&As[0][r0 + p * 32][c0] = ra[p];
            *(uint4*)&Bs[0][r0 + p * 32][c0] = rb[p];
        }
    }
#else
    if (wid == 0) {
        tdm_load_tile_f16(lds_addr_of(&As[0][0][0]), Ag, BK, BM, (unsigned)K, (unsigned)K);
        tdm_load_tile_f16(lds_addr_of(&Bs[0][0][0]), Bg, BK, BN, (unsigned)K, (unsigned)K);
    }
#endif

    for (int kt = 0; kt < steps; ++kt) {
        const int cur = kt & 1;

#if USE_TDM
        if (wid == 0) __builtin_amdgcn_s_wait_tensorcnt(0);
        __syncthreads();
        if (wid == 0 && kt + 1 < steps) {
            const _Float16* An = Ag + (size_t)(kt + 1) * BK;
            const _Float16* Bn = Bg + (size_t)(kt + 1) * BK;
            tdm_load_tile_f16(lds_addr_of(&As[1 - cur][0][0]), An, BK, BM, (unsigned)K, (unsigned)K);
            tdm_load_tile_f16(lds_addr_of(&Bs[1 - cur][0][0]), Bn, BK, BN, (unsigned)K, (unsigned)K);
        }
#else
        __syncthreads();
        if (kt + 1 < steps) {
#pragma unroll
            for (int p = 0; p < 4; ++p) {
                const size_t off = (size_t)(r0 + p * 32) * K + (size_t)(kt + 1) * BK + c0;
                ra[p] = *(const uint4*)(Ag + off);
                rb[p] = *(const uint4*)(Bg + off);
            }
        }
#endif

        // ---- 2 x 8 WMMAs on current LDS buffer ----
#pragma unroll
        for (int kk = 0; kk < 2; ++kk) {
            v16h af[2], bf[4];
#pragma unroll
            for (int i = 0; i < 2; ++i) {
                const _Float16* p = &As[cur][wm + i * 16 + fr][kk * 32];
                v16h v;
                ((float4*)&v)[0] = *(const float4*)(p + fhA);
                ((float4*)&v)[1] = *(const float4*)(p + fhA + 16);
                af[i] = v;
            }
#pragma unroll
            for (int j = 0; j < 4; ++j) {
                const _Float16* p = &Bs[cur][wn + j * 16 + fr][kk * 32];
                v16h v;
                ((float4*)&v)[0] = *(const float4*)(p + fhB);
                ((float4*)&v)[1] = *(const float4*)(p + fhB + 16);
                bf[j] = v;
            }
#pragma unroll
            for (int i = 0; i < 2; ++i)
#pragma unroll
                for (int j = 0; j < 4; ++j)
                    acc[i][j] = __builtin_amdgcn_wmma_f32_16x16x32_f16(
                        false, af[i], false, bf[j], (short)0, acc[i][j], false, false);
        }

#if !USE_TDM
        __syncthreads();
        if (kt + 1 < steps) {
            const int nxt = 1 - cur;
#pragma unroll
            for (int p = 0; p < 4; ++p) {
                *(uint4*)&As[nxt][r0 + p * 32][c0] = ra[p];
                *(uint4*)&Bs[nxt][r0 + p * 32][c0] = rb[p];
            }
        }
#endif
    }

    // ---- epilogue: lane l, vgpr v -> row = v + 8*(l>>4), col = l&15 ----
    const int rsub = (lane >> 4) * 8;
    const int csub = lane & 15;
#pragma unroll
    for (int i = 0; i < 2; ++i) {
#pragma unroll
        for (int j = 0; j < 4; ++j) {
#pragma unroll
            for (int v = 0; v < 8; ++v) {
                const size_t row = m0 + wm + i * 16 + rsub + v;
                const int    col = (int)n0 + wn + j * 16 + csub;
                const float  val = acc[i][j][v];
                if (EPI == 0) {
                    outH[row * FC_OUT + col] = (_Float16)(val + bias0[col]);
                } else {
                    if (row < N_ROIS) {
                        if (col < N_LOC)
                            outLoc[row * N_LOC + col] = val + bias0[col];
                        else if (col < NH)
                            outScore[row * N_SCORE + (col - N_LOC)] =
                                val + bias1[col - N_LOC];
                    }
                }
            }
        }
    }
}

// ============================================================
extern "C" void kernel_launch(void* const* d_in, const int* in_sizes, int n_in,
                              void* d_out, int out_size, void* d_ws, size_t ws_size,
                              hipStream_t stream) {
    const float* feature = (const float*)d_in[0];
    const float* rois    = (const float*)d_in[1];
    const int*   ih      = (const int*)d_in[2];
    const int*   iw      = (const int*)d_in[3];
    const float* W_fc    = (const float*)d_in[4];
    const float* b_fc    = (const float*)d_in[5];
    const float* W_loc   = (const float*)d_in[6];
    const float* b_loc   = (const float*)d_in[7];
    const float* W_score = (const float*)d_in[8];
    const float* b_score = (const float*)d_in[9];

    char* ws = (char*)d_ws;
    _Float16* A_h  = (_Float16*)ws;
    _Float16* Wt_h = (_Float16*)(ws + SZ_A);
    _Float16* fc_h = (_Float16*)(ws + SZ_A + SZ_WT);
    _Float16* Wh_h = (_Float16*)(ws + SZ_A + SZ_WT + SZ_FC);

    float* out_loc   = (float*)d_out;
    float* out_score = out_loc + (size_t)N_ROIS * N_LOC;

    pool_kernel<<<M_PAD, 256, 0, stream>>>(feature, rois, ih, iw, A_h);

    convert_wfc<<<dim3(FC_OUT / 32, FC_IN / 32), dim3(32, 8), 0, stream>>>(W_fc, Wt_h);

    prep_wh<<<NH_PAD, 256, 0, stream>>>(W_loc, W_score, Wh_h);

    gemm_f16<0><<<dim3(M_PAD / 128, FC_OUT / 128), 256, 0, stream>>>(
        A_h, Wt_h, FC_IN, b_fc, nullptr, fc_h, nullptr, nullptr);

    gemm_f16<1><<<dim3(M_PAD / 128, 1), 256, 0, stream>>>(
        fc_h, Wh_h, FC_OUT, b_loc, b_score, nullptr, out_loc, out_score);

    (void)in_sizes; (void)n_in; (void)out_size; (void)ws_size;
}
